// HeteroBatchNorm_88347477279067
// MI455X (gfx1250) — matrix operand order repaired and checked by reading, not compiled, not run on hardware
//
#include <hip/hip_runtime.h>
#include <hip/hip_bf16.h>

typedef __attribute__((ext_vector_type(16))) _Float16 v16h;
typedef __attribute__((ext_vector_type(8)))  float    v8f;

#define NUM_TYPES 32
#define C 64
#define EPS 1e-5f
#define ST_ROWS 128            // rows per super-tile (4 WMMA subtiles)

// ---------------------------------------------------------------------------
// Workspace layout (floats):
//   [0      , 2048) sums     [32][64]
//   [2048   , 4096) sumsq    [32][64]
//   [4096   , 4128) counts   [32]
//   [4128   , 6176) scale    [32][64]   (= invstd * weight)
//   [6176   , 8224) shift    [32][64]   (= bias - mean * scale)
// ---------------------------------------------------------------------------
#define WS_SUMS   0
#define WS_SUMSQ  2048
#define WS_CNT    4096
#define WS_SCALE  4128
#define WS_SHIFT  6176

__global__ void hbn_zero_ws(float* ws) {
    for (int i = threadIdx.x + blockIdx.x * blockDim.x; i < WS_SCALE;
         i += blockDim.x * gridDim.x)
        ws[i] = 0.0f;
}

// ---------------------------------------------------------------------------
// Pass 1: segment sum + sum-of-squares via WMMA (onehot(types)^T @ x).
// Workgroup = 128 threads = 4 waves; wave w owns channels [16w, 16w+16).
// Super-tile = 128 rows staged once in LDS; 4 unguarded WMMA subtiles.
// OOB rows carry type = -1 => their one-hot A column is all-zero, so the
// x load only needs an address clamp, never a branch.
// ---------------------------------------------------------------------------
__global__ void hbn_reduce(const float* __restrict__ x,
                           const int*   __restrict__ types,
                           float* __restrict__ ws, int N) {
    __shared__ int   s_types[ST_ROWS];
    __shared__ float s_cnt[NUM_TYPES];

    const int tid  = threadIdx.x;
    const int wave = tid >> 5;           // 0..3
    const int lane = tid & 31;
    const int c0   = wave * 16;

    if (tid < NUM_TYPES) s_cnt[tid] = 0.0f;

    v8f acc_s0 = {}, acc_s1 = {}, acc_q0 = {}, acc_q1 = {};

    const int m    = lane & 15;            // A row (type within half) / B col
    const int koff = (lane < 16) ? 0 : 8;  // A-matrix K offset for this lane
    const int rs   = (lane < 16) ? 0 : 16; // B-matrix row offset for this lane
    const int col  = c0 + m;               // channel this lane streams
    const int Nm1  = N - 1;

    const int nst = (N + ST_ROWS - 1) / ST_ROWS;
    for (int st = blockIdx.x; st < nst; st += gridDim.x) {
        const int r0 = st * ST_ROWS;
        __syncthreads();                   // protect s_types reuse
        {
            const int r = r0 + tid;
            const int t = (r < N) ? types[r] : -1;
            s_types[tid] = t;
            // branch-free count tally (adds 0.0 for invalid rows)
            atomicAdd(&s_cnt[t & (NUM_TYPES - 1)], (t >= 0) ? 1.0f : 0.0f);
        }
        __syncthreads();

#pragma unroll
        for (int sub = 0; sub < 4; ++sub) {
            const int tb = sub * 32;
            v16h a0, a1, b, b2;
#pragma unroll
            for (int e = 0; e < 16; ++e) {
                // A layout (16-bit A 16x32): element e -> K = e (+8 if e>=8) + koff
                const int K = e + ((e < 8) ? 0 : 8) + koff;
                const int t = s_types[tb + K];
                a0[e] = (_Float16)((t == m)      ? 1.0f : 0.0f);
                a1[e] = (_Float16)((t == m + 16) ? 1.0f : 0.0f);
                // B layout (16-bit B 32x16): element e -> row K (same as A), column col
                int row = r0 + tb + K;
                row = (row < Nm1) ? row : Nm1;         // clamp, no branch
                const float v = x[(size_t)row * C + col];
                b[e]  = (_Float16)v;
                b2[e] = (_Float16)(v * v);
            }
            acc_s0 = __builtin_amdgcn_wmma_f32_16x16x32_f16(false, a0, false, b,
                                                            (short)0, acc_s0, false, false);
            acc_s1 = __builtin_amdgcn_wmma_f32_16x16x32_f16(false, a1, false, b,
                                                            (short)0, acc_s1, false, false);
            acc_q0 = __builtin_amdgcn_wmma_f32_16x16x32_f16(false, a0, false, b2,
                                                            (short)0, acc_q0, false, false);
            acc_q1 = __builtin_amdgcn_wmma_f32_16x16x32_f16(false, a1, false, b2,
                                                            (short)0, acc_q1, false, false);
        }
    }

    // Flush: C/D layout -> VGPR v holds M = v (+8 for lanes 16-31), N = lane&15
    {
        const int mbase = (lane >> 4) * 8;
        const int ch    = c0 + (lane & 15);
        float* sums  = ws + WS_SUMS;
        float* sumsq = ws + WS_SUMSQ;
#pragma unroll
        for (int v = 0; v < 8; ++v) {
            const int mm = mbase + v;
            atomicAdd(&sums [(mm)      * C + ch], acc_s0[v]);
            atomicAdd(&sums [(mm + 16) * C + ch], acc_s1[v]);
            atomicAdd(&sumsq[(mm)      * C + ch], acc_q0[v]);
            atomicAdd(&sumsq[(mm + 16) * C + ch], acc_q1[v]);
        }
    }

    __syncthreads();
    if (tid < NUM_TYPES)
        atomicAdd(&ws[WS_CNT + tid], s_cnt[tid]);
}

// ---------------------------------------------------------------------------
// Pass 2: fold mean/var/weight/bias into per-(type,channel) scale & shift.
// ---------------------------------------------------------------------------
__global__ void hbn_finalize(const float* __restrict__ weight,
                             const float* __restrict__ bias,
                             float* __restrict__ ws) {
    const int i = blockIdx.x * blockDim.x + threadIdx.x;
    if (i >= NUM_TYPES * C) return;
    const int t = i >> 6;

    float cnt = ws[WS_CNT + t];
    cnt = (cnt < 1.0f) ? 1.0f : cnt;
    const float mean = ws[WS_SUMS + i] / cnt;
    float var = ws[WS_SUMSQ + i] / cnt - mean * mean;
    var = (var < 0.0f) ? 0.0f : var;
    const float inv = rsqrtf(var + EPS);
    const float sc  = inv * weight[i];
    ws[WS_SCALE + i] = sc;
    ws[WS_SHIFT + i] = bias[i] - mean * sc;
}

// ---------------------------------------------------------------------------
// Pass 3: out = x * scale[type] + shift[type], float4 wide, tables in LDS.
// ---------------------------------------------------------------------------
__global__ void hbn_apply(const float* __restrict__ x,
                          const int*   __restrict__ types,
                          const float* __restrict__ ws,
                          float* __restrict__ out, int N) {
    __shared__ float s_scale[NUM_TYPES * C];
    __shared__ float s_shift[NUM_TYPES * C];
    for (int i = threadIdx.x; i < NUM_TYPES * C; i += blockDim.x) {
        s_scale[i] = ws[WS_SCALE + i];
        s_shift[i] = ws[WS_SHIFT + i];
    }
    __syncthreads();

    const float4* __restrict__ x4 = (const float4*)x;
    float4* __restrict__ o4 = (float4*)out;
    const size_t total  = (size_t)N * (C / 4);
    const size_t stride = (size_t)gridDim.x * blockDim.x;
    for (size_t idx = (size_t)blockIdx.x * blockDim.x + threadIdx.x;
         idx < total; idx += stride) {
        const int row = (int)(idx >> 4);       // 16 float4 per row
        const int cq  = ((int)idx & 15) << 2;  // channel base
        const int t   = types[row];
        const int bI  = t * C + cq;
        const float4 v = x4[idx];
        float4 r;
        r.x = v.x * s_scale[bI + 0] + s_shift[bI + 0];
        r.y = v.y * s_scale[bI + 1] + s_shift[bI + 1];
        r.z = v.z * s_scale[bI + 2] + s_shift[bI + 2];
        r.w = v.w * s_scale[bI + 3] + s_shift[bI + 3];
        o4[idx] = r;
    }
}

// ---------------------------------------------------------------------------
extern "C" void kernel_launch(void* const* d_in, const int* in_sizes, int n_in,
                              void* d_out, int out_size, void* d_ws, size_t ws_size,
                              hipStream_t stream) {
    const float* x      = (const float*)d_in[0];
    const int*   types  = (const int*)  d_in[1];
    const float* weight = (const float*)d_in[2];
    const float* bias   = (const float*)d_in[3];
    float* out = (float*)d_out;
    float* ws  = (float*)d_ws;

    const int N = in_sizes[0] / C;

    // Zero the accumulation region of the workspace (harness poisons it).
    hbn_zero_ws<<<16, 256, 0, stream>>>(ws);

    // Pass 1: WMMA-based segment sum / sumsq / counts.
    const int nst = (N + ST_ROWS - 1) / ST_ROWS;
    int rblocks = nst < 1024 ? nst : 1024;
    if (rblocks < 1) rblocks = 1;
    hbn_reduce<<<rblocks, 128, 0, stream>>>(x, types, ws, N);

    // Pass 2: finalize scale/shift (2048 elements).
    hbn_finalize<<<(NUM_TYPES * C + 255) / 256, 256, 0, stream>>>(weight, bias, ws);

    // Pass 3: apply.
    const size_t total4 = (size_t)N * (C / 4);
    int ablocks = (int)((total4 + 255) / 256);
    if (ablocks > 4096) ablocks = 4096;
    if (ablocks < 1) ablocks = 1;
    hbn_apply<<<ablocks, 256, 0, stream>>>(x, types, ws, out, N);
}
